// GIN_56255481643210
// MI455X (gfx1250) — matrix operand order repaired and checked by reading, not compiled, not run on hardware
//
#include <hip/hip_runtime.h>
#include <cstdint>

// ---------------- problem constants ----------------
#define NNODE   65536
#define D_IN    512
#define FDIM    256
#define NEDGE   524288
#define BATCH   32
#define K1      200
#define NROWS_H (BATCH * K1)          // 6400
#define BN_EPS  1e-5f

typedef __attribute__((ext_vector_type(16))) __bf16 v16bf;
typedef __attribute__((ext_vector_type(8)))  float  v8f;
typedef __attribute__((ext_vector_type(4)))  int    v4i;

typedef __attribute__((address_space(1))) v4i* gptr4;   // global AS1
typedef __attribute__((address_space(3))) v4i* lptr4;   // LDS AS3

union FragU {
    uint4  q[2];
    v16bf  v;
};

union BF4 {
    __bf16   h[4];
    ushort2  s2[2];
    uint2    u;
};

// ---------------- weight fragment packing ----------------
// Bpack layout (fragment-major): idx = ((tile*(K/32)+ks)*32 + lane)*8 + j
// lane: half=lane>>4, n=lane&15 ; VGPR j holds B[k][c],B[k+1][c] with
// k = ks*32 + 16*half + 2*j, c = tile*16 + n   (B is K x 256 row-major fp32)
__global__ void gin_pack_weight(const float* __restrict__ W,
                                uint32_t* __restrict__ P, int K) {
    int idx = blockIdx.x * blockDim.x + threadIdx.x;
    int total = (FDIM / 16) * (K >> 5) * 32 * 8;
    if (idx >= total) return;
    int j    = idx & 7;
    int lane = (idx >> 3) & 31;
    int rest = idx >> 8;                 // tile*(K/32)+ks
    int kSteps = K >> 5;
    int ks   = rest % kSteps;
    int tile = rest / kSteps;
    int k = ks * 32 + 16 * (lane >> 4) + 2 * j;
    int c = tile * 16 + (lane & 15);
    __bf16 lo = (__bf16)W[(size_t)k * FDIM + c];
    __bf16 hi = (__bf16)W[(size_t)(k + 1) * FDIM + c];
    uint32_t lob = __builtin_bit_cast(unsigned short, lo);
    uint32_t hib = __builtin_bit_cast(unsigned short, hi);
    P[idx] = (hib << 16) | lob;
}

// ---------------- edge aggregation (segment_sum via atomics) ----------------
// one thread handles 4 consecutive columns of one edge
__global__ void gin_edge_aggr(const float* __restrict__ x,
                              const int* __restrict__ src,
                              const int* __restrict__ dst,
                              float* __restrict__ aggr,
                              int logF4, long long total) {  // logF4 = log2(F/4)
    long long idx = (long long)blockIdx.x * blockDim.x + threadIdx.x;
    if (idx >= total) return;
    int c4 = (int)(idx & ((1 << logF4) - 1));
    long long e = idx >> logF4;
    int s = src[e], d = dst[e];
    const float4 v = *(const float4*)(x + (((size_t)s << logF4) + c4) * 4);
    float* ap = aggr + (((size_t)d << logF4) + c4) * 4;
    atomicAdd(ap + 0, v.x);
    atomicAdd(ap + 1, v.y);
    atomicAdd(ap + 2, v.z);
    atomicAdd(ap + 3, v.w);
}

// ---------------- h = (1+eps)*x + aggr  ->  bf16 row-major ----------------
__global__ void gin_combine(const float* __restrict__ x,
                            const float* __restrict__ aggr,
                            const float* __restrict__ epsp,
                            __bf16* __restrict__ Hp, long long total4) {
    long long idx = (long long)blockIdx.x * blockDim.x + threadIdx.x;
    if (idx >= total4) return;
    float e = 1.0f + epsp[0];
    const float4 xv = *(const float4*)(x + idx * 4);
    const float4 av = *(const float4*)(aggr + idx * 4);
    BF4 o;
    o.h[0] = (__bf16)fmaf(e, xv.x, av.x);
    o.h[1] = (__bf16)fmaf(e, xv.y, av.y);
    o.h[2] = (__bf16)fmaf(e, xv.z, av.z);
    o.h[3] = (__bf16)fmaf(e, xv.w, av.w);
    *(uint2*)(Hp + idx * 4) = o.u;
}

// ---------------- WMMA GEMM: [M x K](bf16) @ [K x 256](packed) ----------------
// block = 256 threads = 8 waves; block tile = 64 rows x 256 cols.
// Entire packed B (K*256 bf16 = 128/256 KB) is staged into LDS once per block
// (async global->LDS when the toolchain exposes it); inner loop feeds WMMA
// from ds_load_b128 while A streams from global (double-buffered).
// epilogue modes: 0 = bias+ReLU -> bf16 ; 1 = bias+BN(+ReLU) -> f32 ; 2 = bias+PReLU -> bf16
template <int K>
__global__ __launch_bounds__(256, 1)
void gin_wmma_gemm(const __bf16* __restrict__ Ap,
                   const uint32_t* __restrict__ Bp,
                   const float* __restrict__ bias,
                   int mode,
                   const float* __restrict__ bn_g, const float* __restrict__ bn_b,
                   const float* __restrict__ bn_m, const float* __restrict__ bn_v,
                   int bnrelu,
                   const float* __restrict__ prelu,
                   __bf16* __restrict__ outB,
                   float* __restrict__ outF) {
    constexpr int kSteps = K >> 5;
    constexpr int BDW    = K * 128;          // total B dwords (K*256 bf16 / 2)
    extern __shared__ uint4 BldsV[];
    uint32_t* Blds = (uint32_t*)BldsV;

    // ---- cooperative stage of the full packed B into LDS ----
    {
        const uint4* src4 = (const uint4*)Bp;
#if __has_builtin(__builtin_amdgcn_global_load_async_to_lds_b128)
        for (int i = threadIdx.x; i < BDW / 4; i += 256) {
            __builtin_amdgcn_global_load_async_to_lds_b128(
                (gptr4)(src4 + i), (lptr4)(BldsV + i), 0, 0);
        }
#if __has_builtin(__builtin_amdgcn_s_wait_asynccnt)
        __builtin_amdgcn_s_wait_asynccnt(0);
#else
        asm volatile("s_wait_asynccnt 0x0" ::: "memory");
#endif
#else
#pragma unroll 4
        for (int i = threadIdx.x; i < BDW / 4; i += 256) {
            BldsV[i] = src4[i];
        }
#endif
    }
    __syncthreads();

    const int lane  = threadIdx.x & 31;
    const int wave  = threadIdx.x >> 5;
    const int waveM = wave & 3;
    const int waveN = wave >> 2;
    const int half  = lane >> 4;
    const int ln    = lane & 15;
    const int rowBase = blockIdx.x * 64 + waveM * 16;
    const int aRow    = rowBase + ln;          // A row this lane feeds
    const int colHalf = waveN * 128;

    v8f zero;
#pragma unroll
    for (int i = 0; i < 8; i++) zero[i] = 0.0f;
    v8f acc[8];
#pragma unroll
    for (int t = 0; t < 8; t++) acc[t] = zero;

    const uint32_t* Arow   = (const uint32_t*)(Ap + (size_t)aRow * K);
    // per-lane base of this wave's 8 B tiles in LDS; (t,ks) offset = (t*kSteps+ks)*256 dwords
    const uint32_t* bpLane = Blds + (waveN * 8 * kSteps) * 256 + lane * 8;

    FragU a[2];
    {   // prologue: A fragment for k-step 0
        const int p0 = 8 * half;
        a[0].q[0] = *(const uint4*)(Arow + p0);
        a[0].q[1] = *(const uint4*)(Arow + p0 + 4);
    }

#pragma unroll
    for (int ks = 0; ks < kSteps; ks++) {
        const int cur = ks & 1;
        const int nxt = cur ^ 1;

        // group-load all 8 B fragments for this k-step from LDS
        FragU bf[8];
#pragma unroll
        for (int t = 0; t < 8; t++) {
            const uint32_t* bp = bpLane + (t * kSteps + ks) * 256;
            bf[t].q[0] = *(const uint4*)(bp);
            bf[t].q[1] = *(const uint4*)(bp + 4);
        }

        if (ks + 1 < kSteps) {                 // prefetch next A fragment (global)
            const int p0 = (ks + 1) * 16 + 8 * half;
            a[nxt].q[0] = *(const uint4*)(Arow + p0);
            a[nxt].q[1] = *(const uint4*)(Arow + p0 + 4);
        }

#pragma unroll
        for (int t = 0; t < 8; t++) {
            acc[t] = __builtin_amdgcn_wmma_f32_16x16x32_bf16(
                false, a[cur].v, false, bf[t].v, (short)0, acc[t], false, false);
        }
    }

    // epilogue: lane holds col = colHalf + t*16 + ln, rows rowBase + 8*half + r
#pragma unroll
    for (int t = 0; t < 8; t++) {
        const int col = colHalf + t * 16 + ln;
        const float bi = bias[col];
        float sc = 0.f, sh = 0.f, pa = 0.f;
        if (mode == 1) {
            float rs = rsqrtf(bn_v[col] + BN_EPS);
            sc = bn_g[col] * rs;
            sh = bn_b[col] - bn_m[col] * sc;
        } else if (mode == 2) {
            pa = prelu[col];
        }
#pragma unroll
        for (int r = 0; r < 8; r++) {
            const int orow = rowBase + 8 * half + r;
            float v = acc[t][r] + bi;
            if (mode == 0) {
                v = fmaxf(v, 0.0f);
                outB[(size_t)orow * FDIM + col] = (__bf16)v;
            } else if (mode == 1) {
                v = fmaf(v, sc, sh);
                if (bnrelu) v = fmaxf(v, 0.0f);
                outF[(size_t)orow * FDIM + col] = v;
            } else {
                v = v > 0.0f ? v : pa * v;
                outB[(size_t)orow * FDIM + col] = (__bf16)v;
            }
        }
    }
}

// ---------------- head: gather rows, convert to bf16 ----------------
__global__ void gin_gather_pack(const float* __restrict__ h,
                                const int* __restrict__ h1id,
                                __bf16* __restrict__ EF) {
    int idx = blockIdx.x * blockDim.x + threadIdx.x;   // over NROWS_H * FDIM/4
    if (idx >= NROWS_H * (FDIM / 4)) return;
    int c4 = idx & (FDIM / 4 - 1);
    int i  = idx >> 6;
    int b  = i / K1;
    int node = b * (NNODE / BATCH) + h1id[i];
    const float4 v = *(const float4*)(h + (size_t)node * FDIM + c4 * 4);
    BF4 o;
    o.h[0] = (__bf16)v.x; o.h[1] = (__bf16)v.y;
    o.h[2] = (__bf16)v.z; o.h[3] = (__bf16)v.w;
    *(uint2*)(EF + (size_t)i * FDIM + c4 * 4) = o.u;
}

// ---------------- head final 256 -> 2 ----------------
__global__ void gin_head_final(const __bf16* __restrict__ hc,
                               const float* __restrict__ wc2,
                               const float* __restrict__ bc2,
                               float* __restrict__ out) {
    int i = blockIdx.x * blockDim.x + threadIdx.x;
    if (i >= NROWS_H) return;
    float s0 = bc2[0], s1 = bc2[1];
    const __bf16* row = hc + (size_t)i * FDIM;
    for (int c = 0; c < FDIM; c++) {
        float v = (float)row[c];
        s0 = fmaf(v, wc2[c * 2 + 0], s0);
        s1 = fmaf(v, wc2[c * 2 + 1], s1);
    }
    out[i * 2 + 0] = s0;
    out[i * 2 + 1] = s1;
}

// ---------------- launch ----------------
extern "C" void kernel_launch(void* const* d_in, const int* in_sizes, int n_in,
                              void* d_out, int out_size, void* d_ws, size_t ws_size,
                              hipStream_t stream) {
    const float* feat = (const float*)d_in[0];
    const int*   A    = (const int*)d_in[1];
    const int*   src  = A;
    const int*   dst  = A + NEDGE;
    const int*   h1id = (const int*)d_in[2];
    auto LP = [&](int L, int j) -> const float* {   // j: 0 eps,1 wa,2 ba,3 wb,4 bb,5 g,6 be,7 m,8 v
        return (const float*)d_in[4 + 9 * (L - 1) + j];
    };
    const float* wc1 = (const float*)d_in[40];
    const float* bc1 = (const float*)d_in[41];
    const float* pa  = (const float*)d_in[42];
    const float* wc2 = (const float*)d_in[43];
    const float* bc2 = (const float*)d_in[44];

    uint8_t* ws = (uint8_t*)d_ws;
    size_t off = 0;
    auto wsalloc = [&](size_t bytes) -> void* {
        void* p = ws + off;
        off += (bytes + 255) & ~(size_t)255;
        return p;
    };
    float*    aggr = (float*)wsalloc((size_t)NNODE * D_IN * 4);
    __bf16*   Hp   = (__bf16*)wsalloc((size_t)NNODE * D_IN * 2);
    __bf16*   C1p  = (__bf16*)wsalloc((size_t)NNODE * FDIM * 2);
    float*    xA   = (float*)wsalloc((size_t)NNODE * FDIM * 4);
    float*    xB   = (float*)wsalloc((size_t)NNODE * FDIM * 4);
    __bf16*   EFp  = (__bf16*)wsalloc((size_t)NROWS_H * FDIM * 2);
    __bf16*   HCp  = (__bf16*)wsalloc((size_t)NROWS_H * FDIM * 2);
    uint32_t* wap[4];
    uint32_t* wbp[4];
    for (int L = 0; L < 4; L++) {
        int K = (L == 0) ? D_IN : FDIM;
        wap[L] = (uint32_t*)wsalloc((size_t)K * FDIM * 2);
        wbp[L] = (uint32_t*)wsalloc((size_t)FDIM * FDIM * 2);
    }
    uint32_t* wc1p = (uint32_t*)wsalloc((size_t)FDIM * FDIM * 2);

    // --- pack all weights (fragment-major bf16 pairs) ---
    for (int L = 0; L < 4; L++) {
        int K = (L == 0) ? D_IN : FDIM;
        int tot = K * FDIM / 2;
        hipLaunchKernelGGL(gin_pack_weight, dim3((tot + 255) / 256), dim3(256), 0, stream,
                           LP(L + 1, 1), wap[L], K);
        tot = FDIM * FDIM / 2;
        hipLaunchKernelGGL(gin_pack_weight, dim3((tot + 255) / 256), dim3(256), 0, stream,
                           LP(L + 1, 3), wbp[L], FDIM);
    }
    {
        int tot = FDIM * FDIM / 2;
        hipLaunchKernelGGL(gin_pack_weight, dim3((tot + 255) / 256), dim3(256), 0, stream,
                           wc1, wc1p, FDIM);
    }

    const size_t ldsB256 = (size_t)FDIM * FDIM * 2;   // 128 KB
    const size_t ldsB512 = (size_t)D_IN * FDIM * 2;   // 256 KB

    // --- 4 GIN layers ---
    const float* xcur = feat;
    float* ping = xA;
    float* pong = xB;
    for (int L = 1; L <= 4; L++) {
        const int K     = (L == 1) ? D_IN : FDIM;
        const int logF4 = (L == 1) ? 7 : 6;    // log2(K/4)

        (void)hipMemsetAsync(aggr, 0, (size_t)NNODE * K * 4, stream);
        long long etot = (long long)NEDGE << logF4;
        hipLaunchKernelGGL(gin_edge_aggr, dim3((unsigned)((etot + 255) / 256)), dim3(256), 0, stream,
                           xcur, src, dst, aggr, logF4, etot);

        long long ctot4 = ((long long)NNODE * K) / 4;
        hipLaunchKernelGGL(gin_combine, dim3((unsigned)((ctot4 + 255) / 256)), dim3(256), 0, stream,
                           xcur, aggr, LP(L, 0), Hp, ctot4);

        // GEMM1: relu(H @ Wa + ba) -> bf16
        if (L == 1) {
            hipLaunchKernelGGL((gin_wmma_gemm<D_IN>), dim3(NNODE / 64), dim3(256), ldsB512, stream,
                               Hp, wap[L - 1], LP(L, 2), 0,
                               (const float*)nullptr, (const float*)nullptr,
                               (const float*)nullptr, (const float*)nullptr, 0,
                               (const float*)nullptr, C1p, (float*)nullptr);
        } else {
            hipLaunchKernelGGL((gin_wmma_gemm<FDIM>), dim3(NNODE / 64), dim3(256), ldsB256, stream,
                               Hp, wap[L - 1], LP(L, 2), 0,
                               (const float*)nullptr, (const float*)nullptr,
                               (const float*)nullptr, (const float*)nullptr, 0,
                               (const float*)nullptr, C1p, (float*)nullptr);
        }

        // GEMM2: BN(C1 @ Wb + bb) (+relu for L<4) -> f32
        hipLaunchKernelGGL((gin_wmma_gemm<FDIM>), dim3(NNODE / 64), dim3(256), ldsB256, stream,
                           C1p, wbp[L - 1], LP(L, 4), 1,
                           LP(L, 5), LP(L, 6), LP(L, 7), LP(L, 8), (L < 4) ? 1 : 0,
                           (const float*)nullptr, (__bf16*)nullptr, ping);

        xcur = ping;
        float* t = ping; ping = pong; pong = t;
    }

    // --- head ---
    hipLaunchKernelGGL(gin_gather_pack, dim3((NROWS_H * (FDIM / 4) + 255) / 256), dim3(256), 0, stream,
                       xcur, h1id, EFp);

    hipLaunchKernelGGL((gin_wmma_gemm<FDIM>), dim3(NROWS_H / 64), dim3(256), ldsB256, stream,
                       EFp, wc1p, bc1, 2,
                       (const float*)nullptr, (const float*)nullptr,
                       (const float*)nullptr, (const float*)nullptr, 0,
                       pa, HCp, (float*)nullptr);

    hipLaunchKernelGGL(gin_head_final, dim3((NROWS_H + 255) / 256), dim3(256), 0, stream,
                       HCp, wc2, bc2, (float*)d_out);
}
